// MultilayerGRU_73400991089432
// MI455X (gfx1250) — compile-verified, matrix operand order checked by reading, not run on hardware
//
#include <hip/hip_runtime.h>

// Problem: B=64, S=1024, I=H=O=512, L=3.
// Structural facts from the reference's indexing:
//   * states[0] is never updated -> layer-0 recurrent state is constant.
//   * For layers 1,2: x == h      -> z/r x-side and h-side weights merge.
// => every (b,t) row is independent; 13 GEMMs of 65536x512x512 (446 GFLOP),
//    pure throughput problem. bf16 WMMA 16x16x32 with f32 accumulate.
// Memory floor: read X 128MB + write out 128MB ~ 11us @ 23.3TB/s -> compute
// bound; weights (6.5MB bf16) stay hot in the 192MB L2 across all 1024 blocks.

#define B_ 64
#define S_ 1024
#define I_ 512
#define H_ 512
#define O_ 512
#define MATSZ (512 * 512)

typedef __attribute__((ext_vector_type(16))) __bf16 v16bf;
typedef __attribute__((ext_vector_type(8)))  __bf16 v8bf;
typedef __attribute__((ext_vector_type(4)))  __bf16 v4bf;
typedef __attribute__((ext_vector_type(8)))  float  v8f;

__device__ __forceinline__ v8f wmma_bf16(v16bf a, v16bf b, v8f c) {
  return __builtin_amdgcn_wmma_f32_16x16x32_bf16(false, a, false, b, (short)0, c,
                                                 false, false);
}

// Load one 16x32 bf16 operand fragment for this lane.
// Lanes 0-15 hold K=[kk..kk+7],[kk+16..kk+23]; lanes 16-31 the other halves.
__device__ __forceinline__ v16bf frag16(const __bf16* rowptr, int kk, int koff) {
  v16bf f;
  ((v8bf*)&f)[0] = *(const v8bf*)(rowptr + kk + koff);
  ((v8bf*)&f)[1] = *(const v8bf*)(rowptr + kk + koff + 16);
  return f;
}

// ---------------- weight prep: transpose (+optional merge) fp32 -> bf16 -----
__global__ void k_prep(const float* __restrict__ a, const float* __restrict__ b2,
                       __bf16* __restrict__ dst, int hasB) {
  int i = blockIdx.x * 256 + threadIdx.x;          // exactly 512*512 threads
  int n = i >> 9, k = i & 511;
  float v = a[(size_t)k * 512 + n];
  if (hasB) v += b2[(size_t)k * 512 + n];
  dst[(size_t)n * 512 + k] = (__bf16)v;            // store N-major (B-matrix)
}

// ---------------- input conversion: fp32 -> bf16, vectorized ----------------
__global__ void k_convert_x(const float4* __restrict__ x, __bf16* __restrict__ xb) {
  size_t i = (size_t)blockIdx.x * 256 + threadIdx.x;   // 8,388,608 float4s
  float4 v = x[i];
  v4bf o = {(__bf16)v.x, (__bf16)v.y, (__bf16)v.z, (__bf16)v.w};
  *(v4bf*)(xb + 4 * i) = o;
}

// ---------------- layer-0 gate constants: Cz = h0c@Wzh0+bzx0, Cr likewise ---
__global__ void k_gate_consts(const float* __restrict__ hs,
                              const __bf16* __restrict__ WT,
                              const float* __restrict__ bzx,
                              const float* __restrict__ brx,
                              float* __restrict__ Cz, float* __restrict__ Cr) {
  extern __shared__ __bf16 sh[];                   // 64x512 bf16 = 64KB
  const int tid = threadIdx.x, lane = tid & 31, wave = tid >> 5;
  const int l15 = lane & 15;
  const int koff = (lane & 16) ? 8 : 0;
  const int mDhi = (lane & 16) ? 8 : 0;

  for (int i = tid; i < 64 * 512; i += 256) {      // A = hidden_state[:,0,:]
    int m = i >> 9, k = i & 511;
    sh[i] = (__bf16)hs[(size_t)m * 3 * 512 + k];
  }
  __syncthreads();

  const int cb = blockIdx.x * 128 + wave * 16;
  const int colD = cb + l15;
  const __bf16* wz = WT + (size_t)13 * MATSZ + (size_t)(cb + l15) * 512;
  const __bf16* wr = WT + (size_t)14 * MATSZ + (size_t)(cb + l15) * 512;
  const float bz = bzx[colD], br = brx[colD];
  v8f za[4], ra[4];
#pragma unroll
  for (int j = 0; j < 4; ++j)
#pragma unroll
    for (int i = 0; i < 8; ++i) { za[j][i] = bz; ra[j][i] = br; }

#pragma unroll 2
  for (int kk = 0; kk < 512; kk += 32) {
    v16bf bzf = frag16(wz, kk, koff);
    v16bf brf = frag16(wr, kk, koff);
#pragma unroll
    for (int j = 0; j < 4; ++j) {
      v16bf a = frag16(sh + (size_t)(j * 16 + l15) * 512, kk, koff);
      za[j] = wmma_bf16(a, bzf, za[j]);
      ra[j] = wmma_bf16(a, brf, ra[j]);
    }
  }
#pragma unroll
  for (int j = 0; j < 4; ++j)
#pragma unroll
    for (int i = 0; i < 8; ++i) {
      int m = j * 16 + i + mDhi;
      Cz[(size_t)m * 512 + colD] = za[j][i];
      Cr[(size_t)m * 512 + colD] = ra[j][i];
    }
}

// ---------------- frozen layer-0 hidden slice copy --------------------------
__global__ void k_copy_h0(const float* __restrict__ hs, float* __restrict__ out) {
  int i = blockIdx.x * 256 + threadIdx.x;          // exactly 64*512 threads
  int b = i >> 9, h = i & 511;
  out[(size_t)B_ * S_ * O_ + (size_t)b * 3 * 512 + h] =
      hs[(size_t)b * 3 * 512 + h];
}

// ---------------- fused 3-layer GRU + output projection ---------------------
// One block = 64 consecutive (b,t) rows (same b). LDS: x double-buffer, r*h,
// z, plus layer-0 per-batch constants. Weights stream from L2 (hot, 6.5MB).
// x tile is staged with CDNA5 async global->LDS copies (ASYNCcnt).
__global__ void __launch_bounds__(256, 1)
k_gru_fused(const __bf16* __restrict__ Xbf, const float* __restrict__ hs,
            const __bf16* __restrict__ WT,
            const float* __restrict__ bzx, const float* __restrict__ brx,
            const float* __restrict__ bgx, const float* __restrict__ bo,
            const float* __restrict__ Cz, const float* __restrict__ Cr,
            float* __restrict__ out) {
  extern __shared__ __bf16 lds[];
  __bf16* xbuf0 = lds;                       // 64*512 bf16 (LDS offset 0)
  __bf16* xbuf1 = lds + 64 * 512;
  __bf16* rhb   = lds + 2 * 64 * 512;
  __bf16* zst   = lds + 3 * 64 * 512;
  float*  h0c   = (float*)(lds + 4 * 64 * 512);   // 512 f32
  float*  czc   = h0c + 512;
  float*  crc   = czc + 512;

  const int tid = threadIdx.x, lane = tid & 31, wave = tid >> 5;
  const int l15 = lane & 15;
  const int koff = (lane & 16) ? 8 : 0;
  const int mDhi = (lane & 16) ? 8 : 0;

  const int tile = blockIdx.x;               // 0..1023
  const int b  = tile >> 4;
  const int t0 = (tile & 15) * 64;
  const bool lastTile = ((tile & 15) == 15); // contains t = S-1 at m == 63

  // ---- async stage: 64 contiguous bf16 rows (64KB) global -> LDS ----------
  {
    const char* src = (const char*)(Xbf + ((size_t)b * S_ + t0) * I_);
    for (int i = tid; i < 4096; i += 256) {          // 4096 x 16B chunks
      unsigned ldsoff = (unsigned)(i * 16);          // xbuf0 at LDS offset 0
      const void* g = src + (size_t)i * 16;
      asm volatile("global_load_async_to_lds_b128 %0, %1, off"
                   :: "v"(ldsoff), "v"(g) : "memory");
    }
  }
  for (int i = tid; i < 512; i += 256) {
    h0c[i] = hs[(size_t)b * 3 * 512 + i];
    czc[i] = Cz[(size_t)b * 512 + i];
    crc[i] = Cr[(size_t)b * 512 + i];
  }
  asm volatile("s_wait_asynccnt 0" ::: "memory");
  __syncthreads();

  __bf16* xc = xbuf0;
  __bf16* xn = xbuf1;

  for (int layer = 0; layer < 3; ++layer) {
    const bool lay0 = (layer == 0);
    const __bf16* Wz  = WT + (size_t)(layer * 4 + 0) * MATSZ; // merged for l>=1
    const __bf16* Wr  = WT + (size_t)(layer * 4 + 1) * MATSZ;
    const __bf16* Wgx = WT + (size_t)(layer * 4 + 2) * MATSZ;
    const __bf16* Wgh = WT + (size_t)(layer * 4 + 3) * MATSZ;

    // ---------------- z / r gates ----------------
    for (int chunk = 0; chunk < 4; ++chunk) {
      const int cb = chunk * 128 + wave * 16;
      const int colD = cb + l15;
      const __bf16* wzr = Wz + (size_t)(cb + l15) * 512;
      const __bf16* wrr = Wr + (size_t)(cb + l15) * 512;
      if (chunk < 3) {                       // keep the L2 weight stream ahead
        __builtin_prefetch(wzr + 128 * 512, 0, 0);
        __builtin_prefetch(wrr + 128 * 512, 0, 0);
      }
      const float zi = lay0 ? czc[colD] : bzx[layer * 512 + colD];
      const float ri = lay0 ? crc[colD] : brx[layer * 512 + colD];
      v8f zacc[4], racc[4];
#pragma unroll
      for (int j = 0; j < 4; ++j)
#pragma unroll
        for (int i = 0; i < 8; ++i) { zacc[j][i] = zi; racc[j][i] = ri; }

#pragma unroll 2
      for (int kk = 0; kk < 512; kk += 32) {
        v16bf bz = frag16(wzr, kk, koff);
        v16bf br = frag16(wrr, kk, koff);
#pragma unroll
        for (int j = 0; j < 4; ++j) {
          v16bf a = frag16(xc + (size_t)(j * 16 + l15) * 512, kk, koff);
          zacc[j] = wmma_bf16(a, bz, zacc[j]);
          racc[j] = wmma_bf16(a, br, racc[j]);
        }
      }
#pragma unroll
      for (int j = 0; j < 4; ++j)
#pragma unroll
        for (int i = 0; i < 8; ++i) {
          const int m = j * 16 + i + mDhi;
          const float z = 1.f / (1.f + __expf(-zacc[j][i]));
          const float r = 1.f / (1.f + __expf(-racc[j][i]));
          const float hv = lay0 ? h0c[colD] : (float)xc[(size_t)m * 512 + colD];
          rhb[(size_t)m * 512 + colD] = (__bf16)(r * hv);
          zst[(size_t)m * 512 + colD] = (__bf16)z;
        }
    }
    __syncthreads();

    // ---------------- g gate + h update ----------------
    for (int chunk = 0; chunk < 4; ++chunk) {
      const int cb = chunk * 128 + wave * 16;
      const int colD = cb + l15;
      const __bf16* wgxr = Wgx + (size_t)(cb + l15) * 512;
      const __bf16* wghr = Wgh + (size_t)(cb + l15) * 512;
      if (chunk < 3) {
        __builtin_prefetch(wgxr + 128 * 512, 0, 0);
        __builtin_prefetch(wghr + 128 * 512, 0, 0);
      }
      const float gi = bgx[layer * 512 + colD];
      v8f gacc[4];
#pragma unroll
      for (int j = 0; j < 4; ++j)
#pragma unroll
        for (int i = 0; i < 8; ++i) gacc[j][i] = gi;

#pragma unroll 2
      for (int kk = 0; kk < 512; kk += 32) {
        v16bf bx = frag16(wgxr, kk, koff);
        v16bf bh = frag16(wghr, kk, koff);
#pragma unroll
        for (int j = 0; j < 4; ++j) {
          v16bf ax = frag16(xc + (size_t)(j * 16 + l15) * 512, kk, koff);
          gacc[j] = wmma_bf16(ax, bx, gacc[j]);
          v16bf ar = frag16(rhb + (size_t)(j * 16 + l15) * 512, kk, koff);
          gacc[j] = wmma_bf16(ar, bh, gacc[j]);
        }
      }
#pragma unroll
      for (int j = 0; j < 4; ++j)
#pragma unroll
        for (int i = 0; i < 8; ++i) {
          const int m = j * 16 + i + mDhi;
          const float g = tanhf(gacc[j][i]);
          const float z = (float)zst[(size_t)m * 512 + colD];
          const float hv = lay0 ? h0c[colD] : (float)xc[(size_t)m * 512 + colD];
          const float hn = z * hv + (1.f - z) * g;
          xn[(size_t)m * 512 + colD] = (__bf16)hn;
          // hidden_out[:,1,:]=h0(S-1), [:,2,:]=h1(S-1); write in f32.
          if (lastTile && m == 63 && layer < 2)
            out[(size_t)B_ * S_ * O_ + ((size_t)b * 3 + layer + 1) * 512 + colD] = hn;
        }
    }
    __syncthreads();
    __bf16* tmp = xc; xc = xn; xn = tmp;
  }

  // ---------------- output projection ----------------
  const __bf16* Wo = WT + (size_t)12 * MATSZ;
  for (int chunk = 0; chunk < 4; ++chunk) {
    const int cb = chunk * 128 + wave * 16;
    const int colD = cb + l15;
    const __bf16* wor = Wo + (size_t)(cb + l15) * 512;
    if (chunk < 3) __builtin_prefetch(wor + 128 * 512, 0, 0);
    const float oi = bo[colD];
    v8f acc[4];
#pragma unroll
    for (int j = 0; j < 4; ++j)
#pragma unroll
      for (int i = 0; i < 8; ++i) acc[j][i] = oi;

#pragma unroll 2
    for (int kk = 0; kk < 512; kk += 32) {
      v16bf bw = frag16(wor, kk, koff);
#pragma unroll
      for (int j = 0; j < 4; ++j) {
        v16bf a = frag16(xc + (size_t)(j * 16 + l15) * 512, kk, koff);
        acc[j] = wmma_bf16(a, bw, acc[j]);
      }
    }
#pragma unroll
    for (int j = 0; j < 4; ++j)
#pragma unroll
      for (int i = 0; i < 8; ++i) {
        const int m = j * 16 + i + mDhi;
        out[((size_t)b * S_ + (t0 + m)) * O_ + colD] = acc[j][i];
      }
  }
}

// ---------------- host launcher ---------------------------------------------
extern "C" void kernel_launch(void* const* d_in, const int* in_sizes, int n_in,
                              void* d_out, int out_size, void* d_ws, size_t ws_size,
                              hipStream_t stream) {
  (void)in_sizes; (void)n_in; (void)out_size; (void)ws_size;
  const float* X    = (const float*)d_in[0];
  const float* HS   = (const float*)d_in[1];
  const float* Wzx  = (const float*)d_in[2];
  const float* bzx  = (const float*)d_in[3];
  const float* Wzh  = (const float*)d_in[4];
  const float* Wrx  = (const float*)d_in[5];
  const float* brx  = (const float*)d_in[6];
  const float* Wrh  = (const float*)d_in[7];
  const float* Wgx  = (const float*)d_in[8];
  const float* bgx  = (const float*)d_in[9];
  const float* Wgh  = (const float*)d_in[10];
  const float* Wo   = (const float*)d_in[11];
  const float* bo   = (const float*)d_in[12];
  float* out = (float*)d_out;

  // Workspace (~75 MB): 15 transposed bf16 weight mats (7.5MB) +
  // layer-0 gate constants (256KB) + bf16 copy of X (64MB).
  char* ws = (char*)d_ws;
  __bf16* WT = (__bf16*)ws;
  float* Cz = (float*)(ws + (size_t)15 * MATSZ * 2);
  float* Cr = Cz + 64 * 512;
  __bf16* Xbf = (__bf16*)(ws + (size_t)15 * MATSZ * 2 + 2 * 64 * 512 * 4);

  // Weight slots: per layer {Wz(merged for l>=1), Wr(merged), Wgx, Wgh},
  // slot 12 = Wout, 13/14 = Wzh0/Wrh0 (for the layer-0 constants).
  struct Src { const float* a; const float* b; };
  const Src mt[15] = {
      {Wzx, nullptr}, {Wrx, nullptr}, {Wgx, nullptr}, {Wgh, nullptr},
      {Wzx + MATSZ, Wzh + MATSZ}, {Wrx + MATSZ, Wrh + MATSZ},
      {Wgx + MATSZ, nullptr}, {Wgh + MATSZ, nullptr},
      {Wzx + 2 * MATSZ, Wzh + 2 * MATSZ}, {Wrx + 2 * MATSZ, Wrh + 2 * MATSZ},
      {Wgx + 2 * MATSZ, nullptr}, {Wgh + 2 * MATSZ, nullptr},
      {Wo, nullptr}, {Wzh, nullptr}, {Wrh, nullptr}};
  for (int m = 0; m < 15; ++m)
    k_prep<<<1024, 256, 0, stream>>>(mt[m].a, mt[m].b ? mt[m].b : mt[m].a,
                                     WT + (size_t)m * MATSZ, mt[m].b ? 1 : 0);

  k_convert_x<<<32768, 256, 0, stream>>>((const float4*)X, Xbf);
  k_gate_consts<<<4, 256, 65536, stream>>>(HS, WT, bzx, brx, Cz, Cr);
  k_copy_h0<<<128, 256, 0, stream>>>(HS, out);

  // 262144 B activations + 6144 B constants of dynamic LDS (<= 320KB/WGP).
  (void)hipFuncSetAttribute((const void*)k_gru_fused,
                            hipFuncAttributeMaxDynamicSharedMemorySize, 268288);
  k_gru_fused<<<1024, 256, 268288, stream>>>(Xbf, HS, WT, bzx, brx, bgx, bo,
                                             Cz, Cr, out);
}
